// ImprovedGCN_12034498363886
// MI455X (gfx1250) — matrix operand order repaired and checked by reading, not compiled, not run on hardware
//
#include <hip/hip_runtime.h>

// CDNA5 / gfx1250 GCN pipeline. Matrix work on v_wmma_f32_16x16x32_f16.
// Persistent-wave GEMM keeps the 64x64 weight matrix resident in VGPRs.

typedef __attribute__((ext_vector_type(16))) _Float16 v16h;
typedef __attribute__((ext_vector_type(8)))  float    v8f;

#define NNODES 100000
#define NEDGES 1600000
#define NGRAPH 512

// ---------------- elementwise helpers ----------------

__global__ void k_fill(float* __restrict__ p, float v, int n) {
  int i = blockIdx.x * blockDim.x + threadIdx.x;
  if (i < n) p[i] = v;
}

__global__ void k_deg_edges(const int* __restrict__ dst, float* __restrict__ deg, int E) {
  int e = blockIdx.x * blockDim.x + threadIdx.x;
  if (e < E) atomicAdd(&deg[dst[e]], 1.0f);
}

__global__ void k_rsqrt_inplace(float* __restrict__ d, int n) {
  int i = blockIdx.x * blockDim.x + threadIdx.x;
  if (i < n) d[i] = rsqrtf(d[i]);   // deg >= 1 always (self loop)
}

__global__ void k_cvt16(_Float16* __restrict__ o, const float* __restrict__ x, int n) {
  int i = blockIdx.x * blockDim.x + threadIdx.x;
  if (i < n) o[i] = (_Float16)x[i];
}

// W is K x Nc row-major f32; Wt becomes Nc x K row-major f16 (i.e. B col-major)
__global__ void k_tcvt(_Float16* __restrict__ Wt, const float* __restrict__ W, int K, int Nc) {
  int i = blockIdx.x * blockDim.x + threadIdx.x;
  if (i >= K * Nc) return;
  int k = i / Nc, n = i - k * Nc;
  Wt[n * K + k] = (_Float16)W[i];
}

// norm[e] = dinv[src]*dinv[dst]
__global__ void k_norm(const int* __restrict__ src, const int* __restrict__ dst,
                       const float* __restrict__ dinv, float* __restrict__ nrm, int E) {
  int e = blockIdx.x * blockDim.x + threadIdx.x;
  if (e < E) nrm[e] = dinv[src[e]] * dinv[dst[e]];
}

// ---------------- WMMA GEMM:  C[M x NT*16] = A[M x 64] * Bt^T ----------------
// A: M x 64 f16 row-major.  Bt: (NT*16) x 64 f16 row-major (B column-major).
// Persistent waves: B + bias live in registers; grid-stride over 16-row tiles.

template<int NT, bool BIAS, bool RELU, bool WF32, bool WF16>
__global__ __launch_bounds__(128) void k_wmma_gemm(
    const _Float16* __restrict__ A,
    const _Float16* __restrict__ Bt,
    const float*    __restrict__ bias,
    float*          __restrict__ Cf32,
    _Float16*       __restrict__ Cf16,
    int M) {
  const int K = 64;
  const int Ncols = NT * 16;
  // force wave id scalar-uniform so tile-loop control stays on the SALU and
  // EXEC remains all-ones around the WMMAs
  int wave = __builtin_amdgcn_readfirstlane(threadIdx.x >> 5);
  int lane = threadIdx.x & 31;
  int mlo  = lane & 15;
  int hi   = lane >> 4;                       // lanes 16-31

  union AB { v16h v; uint4 q[2]; };

  // Weights resident in VGPRs for the whole kernel (NT*2 x v16h = NT*16 VGPRs).
  AB b[NT][2];
#pragma unroll
  for (int nt = 0; nt < NT; ++nt) {
    const _Float16* Brow = Bt + (size_t)(nt * 16 + mlo) * K;
#pragma unroll
    for (int k0 = 0; k0 < 2; ++k0) {
      // B 32x16 layout: halves 0-7 = K k0*32+hi*8..+7 ; halves 8-15 = +16
      b[nt][k0].q[0] = *(const uint4*)(Brow + k0 * 32 + hi * 8);
      b[nt][k0].q[1] = *(const uint4*)(Brow + k0 * 32 + 16 + hi * 8);
    }
  }
  float bv[NT];
  if (BIAS) {
#pragma unroll
    for (int nt = 0; nt < NT; ++nt) bv[nt] = bias[nt * 16 + mlo];
  }

  const int numTiles = M >> 4;
  const int step = gridDim.x * 4;
  for (int tile = blockIdx.x * 4 + wave; tile < numTiles; tile += step) {
    int m0 = tile << 4;
    const _Float16* Arow = A + (size_t)(m0 + mlo) * K;
    AB a[2];
#pragma unroll
    for (int k0 = 0; k0 < 2; ++k0) {
      // A 16x32 layout: halves 0-7 = K k0*32+hi*8..+7 ; halves 8-15 = +16
      a[k0].q[0] = *(const uint4*)(Arow + k0 * 32 + hi * 8);
      a[k0].q[1] = *(const uint4*)(Arow + k0 * 32 + 16 + hi * 8);
    }
    // speculative prefetch of next tile's A row
    __builtin_prefetch(Arow + (size_t)step * 16 * K, 0, 3);

    // base row for this lane's D fragment: lanes16-31 own rows m0+8..m0+15
    const size_t obase = (size_t)(m0 + hi * 8) * Ncols + mlo;

#pragma unroll
    for (int nt = 0; nt < NT; ++nt) {
      v8f c = {};
      c = __builtin_amdgcn_wmma_f32_16x16x32_f16(false, a[0].v, false, b[nt][0].v,
                                                 (short)0, c, false, false);
      c = __builtin_amdgcn_wmma_f32_16x16x32_f16(false, a[1].v, false, b[nt][1].v,
                                                 (short)0, c, false, false);
      float vout[8];
#pragma unroll
      for (int r = 0; r < 8; ++r) {
        float v = c[r];
        if (BIAS) v += bv[nt];
        if (RELU) v = fmaxf(v, 0.0f);
        vout[r] = v;
      }
      if (WF32) {
        float* p = Cf32 + obase + nt * 16;    // row stride folds into IOFFSET
#pragma unroll
        for (int r = 0; r < 8; ++r) p[r * Ncols] = vout[r];
      }
      if (WF16) {
        _Float16* p = Cf16 + obase + nt * 16;
#pragma unroll
        for (int r = 0; r < 8; ++r) p[r * Ncols] = (_Float16)vout[r];
      }
    }
  }
}

// ---------------- edge scatter:  agg[dst] += h[src] * norm ----------------
// One 64-thread block per edge: e is block-uniform so src/dst/norm become
// scalar loads; only the h row read + one atomic per lane hit the vector path.

__global__ __launch_bounds__(64) void k_scatter(const int* __restrict__ src,
                                                const int* __restrict__ dst,
                                                const float* __restrict__ nrm,
                                                const float* __restrict__ h,
                                                float* __restrict__ agg) {
  int e = blockIdx.x;
  int f = threadIdx.x;                        // 0..63
  int s = src[e], d = dst[e];
  float w = nrm[e];
  atomicAdd(&agg[(size_t)d * 64 + f], h[(size_t)s * 64 + f] * w);
}

// self-loop term + bias + relu, write f32 (pool/gather input) and f16 (next GEMM input)
__global__ void k_bias_relu(float* __restrict__ g, const float* __restrict__ agg,
                            const float* __restrict__ dinv, const float* __restrict__ b,
                            _Float16* __restrict__ hf16, int N) {
  int idx = blockIdx.x * blockDim.x + threadIdx.x;
  if (idx >= N * 64) return;
  int i = idx >> 6, f = idx & 63;
  float di = dinv[i];
  float v = agg[idx] + g[idx] * di * di + b[f];
  v = fmaxf(v, 0.0f);
  g[idx]    = v;
  hf16[idx] = (_Float16)v;
}

// ---------------- mean pool ----------------

__global__ void k_pool_sum(const float* __restrict__ g, const int* __restrict__ batch,
                           float* __restrict__ sums, int N) {
  int idx = blockIdx.x * blockDim.x + threadIdx.x;
  if (idx >= N * 64) return;
  int i = idx >> 6, f = idx & 63;
  atomicAdd(&sums[(size_t)batch[i] * 64 + f], g[idx]);
}

__global__ void k_pool_cnt(const int* __restrict__ batch, float* __restrict__ cnts, int N) {
  int i = blockIdx.x * blockDim.x + threadIdx.x;
  if (i < N) atomicAdd(&cnts[batch[i]], 1.0f);
}

__global__ void k_pool_div(_Float16* __restrict__ ph, const float* __restrict__ sums,
                           const float* __restrict__ cnts, int G) {
  int idx = blockIdx.x * blockDim.x + threadIdx.x;
  if (idx >= G * 64) return;
  int gi = idx >> 6;
  ph[idx] = (_Float16)(sums[idx] / fmaxf(cnts[gi], 1.0f));
}

// ---------------- host orchestration ----------------

extern "C" void kernel_launch(void* const* d_in, const int* in_sizes, int n_in,
                              void* d_out, int out_size, void* d_ws, size_t ws_size,
                              hipStream_t stream) {
  const float* x   = (const float*)d_in[0];
  const int*   ei  = (const int*)  d_in[1];   // [2 x E]
  const int*   bat = (const int*)  d_in[2];
  const float* W1  = (const float*)d_in[3];
  const float* b1  = (const float*)d_in[4];
  const float* W2  = (const float*)d_in[5];
  const float* b2  = (const float*)d_in[6];
  const float* l1W = (const float*)d_in[7];
  const float* l1b = (const float*)d_in[8];
  const float* l2W = (const float*)d_in[9];
  const float* l2b = (const float*)d_in[10];

  const int N = NNODES, E = NEDGES, G = NGRAPH;
  const int* srcI = ei;
  const int* dstI = ei + E;

  char* ws = (char*)d_ws;
  size_t off = 0;
  auto carve = [&](size_t bytes) -> void* {
    void* p = ws + off;
    off += (bytes + 255) & ~(size_t)255;     // keep every buffer 256B-aligned
    return p;
  };
  float*    dinv = (float*)   carve((size_t)N * 4);
  float*    nrm  = (float*)   carve((size_t)E * 4);
  _Float16* xh   = (_Float16*)carve((size_t)N * 64 * 2);
  _Float16* W1t  = (_Float16*)carve(64 * 64 * 2);
  _Float16* W2t  = (_Float16*)carve(64 * 64 * 2);
  _Float16* l1t  = (_Float16*)carve(64 * 64 * 2);
  _Float16* l2t  = (_Float16*)carve(16 * 64 * 2);
  float*    g    = (float*)   carve((size_t)N * 64 * 4);   // GEMM out / relu out
  float*    agg  = (float*)   carve((size_t)N * 64 * 4);   // scatter accumulator
  _Float16* hf16 = (_Float16*)carve((size_t)N * 64 * 2);
  float*    sums = (float*)   carve((size_t)G * 64 * 4);
  float*    cnts = (float*)   carve((size_t)G * 4);
  _Float16* ph   = (_Float16*)carve((size_t)G * 64 * 2);
  _Float16* m1h  = (_Float16*)carve((size_t)G * 64 * 2);
  (void)ws_size; (void)in_sizes; (void)n_in; (void)out_size;

  dim3 b256(256);
  auto blks = [](long long n, int b) { return dim3((unsigned)((n + b - 1) / b)); };

  // 1) symmetric normalization: deg = 1 (self loop) + in-degree; dinv = rsqrt(deg)
  k_fill<<<blks(N, 256), b256, 0, stream>>>(dinv, 1.0f, N);
  k_deg_edges<<<blks(E, 256), b256, 0, stream>>>(dstI, dinv, E);
  k_rsqrt_inplace<<<blks(N, 256), b256, 0, stream>>>(dinv, N);
  k_norm<<<blks(E, 256), b256, 0, stream>>>(srcI, dstI, dinv, nrm, E);

  // 2) one-time f16 conversions / weight transposes
  k_cvt16<<<blks((long long)N * 64, 256), b256, 0, stream>>>(xh, x, N * 64);
  k_tcvt<<<blks(64 * 64, 256), b256, 0, stream>>>(W1t, W1, 64, 64);
  k_tcvt<<<blks(64 * 64, 256), b256, 0, stream>>>(W2t, W2, 64, 64);
  k_tcvt<<<blks(64 * 64, 256), b256, 0, stream>>>(l1t, l1W, 64, 64);
  k_tcvt<<<blks(64 * 16, 256), b256, 0, stream>>>(l2t, l2W, 64, 16);

  const int convGrid = 512;                  // persistent waves: 2048 waves / 6250 tiles
  const int mlpGrid  = 8;                    // 32 tiles

  // 3) conv1: h = x@W1 ; scatter ; +self +b1 ; relu
  k_wmma_gemm<4, false, false, true, false>
      <<<convGrid, 128, 0, stream>>>(xh, W1t, nullptr, g, nullptr, N);
  k_fill<<<blks((long long)N * 64, 256), b256, 0, stream>>>(agg, 0.0f, N * 64);
  k_scatter<<<E, 64, 0, stream>>>(srcI, dstI, nrm, g, agg);
  k_bias_relu<<<blks((long long)N * 64, 256), b256, 0, stream>>>(g, agg, dinv, b1, hf16, N);

  // 4) conv2
  k_wmma_gemm<4, false, false, true, false>
      <<<convGrid, 128, 0, stream>>>(hf16, W2t, nullptr, g, nullptr, N);
  k_fill<<<blks((long long)N * 64, 256), b256, 0, stream>>>(agg, 0.0f, N * 64);
  k_scatter<<<E, 64, 0, stream>>>(srcI, dstI, nrm, g, agg);
  k_bias_relu<<<blks((long long)N * 64, 256), b256, 0, stream>>>(g, agg, dinv, b2, hf16, N);

  // 5) global mean pool
  k_fill<<<blks(G * 64, 256), b256, 0, stream>>>(sums, 0.0f, G * 64);
  k_fill<<<blks(G, 256), b256, 0, stream>>>(cnts, 0.0f, G);
  k_pool_sum<<<blks((long long)N * 64, 256), b256, 0, stream>>>(g, bat, sums, N);
  k_pool_cnt<<<blks(N, 256), b256, 0, stream>>>(bat, cnts, N);
  k_pool_div<<<blks(G * 64, 256), b256, 0, stream>>>(ph, sums, cnts, G);

  // 6) MLP head: relu(pooled@lin1+b) @ lin2 + b  -> d_out [512 x 16] f32
  k_wmma_gemm<4, true, true, false, true>
      <<<mlpGrid, 128, 0, stream>>>(ph, l1t, l1b, nullptr, m1h, G);
  k_wmma_gemm<1, true, false, true, false>
      <<<mlpGrid, 128, 0, stream>>>(m1h, l2t, l2b, (float*)d_out, nullptr, G);
}
